// Short_encoder_9268539425022
// MI455X (gfx1250) — matrix operand order repaired and checked by reading, not compiled, not run on hardware
//
#include <hip/hip_runtime.h>
#include <stdint.h>

typedef unsigned short u16;
typedef unsigned int   u32;
typedef __attribute__((ext_vector_type(16))) __bf16 v16bf;
typedef __attribute__((ext_vector_type(8)))  float  v8f;

#define BQ   4
#define SEQ  1024
#define DIM  512
#define NH   8
#define DKH  64
#define DFF  2048
#define NLAY 2
#define BH   (BQ * NH)     // 32
#define BLM  (BQ * SEQ)    // 4096 rows of activations
#define HWIN 6
#define NEGV (-1e9f)
#define SCL  0.125f        // 64^-0.5

#if defined(__AMDGCN__) && __has_builtin(__builtin_amdgcn_global_load_async_to_lds_b128)
#define HAVE_ASYNC_LDS 1
#else
#define HAVE_ASYNC_LDS 0
#endif

union Frag {
  v16bf v;
  u32   u[8];
  uint4 q[2];
};

__device__ __forceinline__ u16 f2bf(float f) {
  u32 u = __float_as_uint(f);
  u32 r = 0x7FFFu + ((u >> 16) & 1u);
  return (u16)((u + r) >> 16);
}

// 16-byte copy global -> LDS. Async (ASYNCcnt, no VGPR round trip) when the
// gfx1250 builtin is available, else plain load/store.
// Builtin signature (from hipcc diagnostic): param0 = GCC-vector int4 pointer
// in addrspace(1), param1 = same in addrspace(3), then imm offset, imm cpol.
typedef int v4i_g __attribute__((vector_size(16)));
typedef v4i_g __attribute__((address_space(1))) as1_v4i;
typedef v4i_g __attribute__((address_space(3))) as3_v4i;

__device__ __forceinline__ void cp_g2l_16(u16* lds_dst, const u16* g_src) {
#if HAVE_ASYNC_LDS
  __builtin_amdgcn_global_load_async_to_lds_b128(
      (as1_v4i*)(uintptr_t)g_src, (as3_v4i*)(uintptr_t)lds_dst, 0, 0);
#else
  *(uint4*)lds_dst = *(const uint4*)g_src;
#endif
}

__device__ __forceinline__ void wait_async0() {
#if HAVE_ASYNC_LDS
#if __has_builtin(__builtin_amdgcn_s_wait_asynccnt)
  __builtin_amdgcn_s_wait_asynccnt(0);
#else
  asm volatile("s_wait_asynccnt 0x0" ::: "memory");
#endif
#endif
}

// A-matrix fragment, 16-bit 16x32 ISA layout:
// lanes 0-15: dwords 0-3 -> K=0..7, dwords 4-7 -> K=16..23 (row M = lane)
// lanes 16-31: K=8..15 and K=24..31
template <int STRIDE>
__device__ __forceinline__ Frag load_fragA(const u16 (*lds)[STRIDE], int row0,
                                           int koff, int lane) {
  Frag f;
  const u16* src = &lds[row0 + (lane & 15)][koff];
  const int kb = (lane >> 4) << 3;
#pragma unroll
  for (int p = 0; p < 4; ++p) {
    f.u[p]     = *(const u32*)(src + kb + 2 * p);
    f.u[4 + p] = *(const u32*)(src + 16 + kb + 2 * p);
  }
  return f;
}

// B-matrix fragment, 16-bit 32x16 ISA layout (LDS holds B transposed: [n][k]):
// lanes 0-15 hold K=0..7,16..23, lanes 16-31 hold K=8..15,24..31; n = lane&15.
template <int STRIDE>
__device__ __forceinline__ Frag load_fragB(const u16 (*lds)[STRIDE], int n0,
                                           int koff, int lane) {
  Frag f;
  const u16* src = &lds[n0 + (lane & 15)][koff + ((lane >> 4) << 3)];
  f.q[0] = *(const uint4*)(src);
  f.q[1] = *(const uint4*)(src + 16);
  return f;
}

// ---------------------------------------------------------------------------
// Generic GEMM: out = act(A[M,K](bf16) @ W[K,N](bf16) + bias), fp32/bf16 out
// Double-buffered LDS; A tile staged via async-to-LDS, B tile transposed
// through VGPRs. Pipeline per k-step: issue stage(next) -> WMMA(cur) ->
// s_wait_asynccnt -> barrier.
// ---------------------------------------------------------------------------
__global__ __launch_bounds__(256) void gemm_bf16(
    const u16* __restrict__ A, const u16* __restrict__ W,
    const float* __restrict__ bias, float* __restrict__ outF,
    u16* __restrict__ outB, int M, int N, int K, int gelu) {
  __shared__ u16 Alds[2][128][64];
  __shared__ u16 Blds[2][128][64];  // [n][k]
  const int t = threadIdx.x;
  const int lane = t & 31;
  const int w = t >> 5;
  const int waveM = w >> 2;  // 0..1 -> 64-row slice
  const int waveN = w & 3;   // 0..3 -> 32-col slice
  const int m0 = blockIdx.y * 128;
  const int n0 = blockIdx.x * 128;

  const v8f vzero = {0.f, 0.f, 0.f, 0.f, 0.f, 0.f, 0.f, 0.f};
  v8f acc[4][2];
#pragma unroll
  for (int i = 0; i < 4; ++i) {
    acc[i][0] = vzero;
    acc[i][1] = vzero;
  }

  const int arow = t >> 1, ahalf = t & 1;    // A: 32 bf16 per thread
  const int bk = t >> 2, bn = (t & 3) * 32;  // B: 32 bf16 per thread (transpose)

  auto stage = [&](int buf, int k0) {
    // A tile [128][64]: contiguous 64B per thread -> 4 async b128
    const u16* gA = A + (size_t)(m0 + arow) * K + k0 + ahalf * 32;
    u16* lA = &Alds[buf][arow][ahalf * 32];
#if HAVE_ASYNC_LDS
    cp_g2l_16(lA, gA);
    cp_g2l_16(lA + 8, gA + 8);
    cp_g2l_16(lA + 16, gA + 16);
    cp_g2l_16(lA + 24, gA + 24);
#else
    {
      const uint4* ga4 = (const uint4*)gA;
      uint4 a0 = ga4[0], a1 = ga4[1], a2 = ga4[2], a3 = ga4[3];
      uint4* la4 = (uint4*)lA;
      la4[0] = a0; la4[1] = a1; la4[2] = a2; la4[3] = a3;
    }
#endif
    // B tile transposed -> Blds[n][k] (needs per-element shuffle, VGPR path)
    const uint4* gb4 = (const uint4*)(W + (size_t)(k0 + bk) * N + n0 + bn);
    uint4 q0 = gb4[0], q1 = gb4[1], q2 = gb4[2], q3 = gb4[3];
    u32 vals[16] = {q0.x, q0.y, q0.z, q0.w, q1.x, q1.y, q1.z, q1.w,
                    q2.x, q2.y, q2.z, q2.w, q3.x, q3.y, q3.z, q3.w};
#pragma unroll
    for (int e = 0; e < 16; ++e) {
      Blds[buf][bn + 2 * e][bk]     = (u16)(vals[e] & 0xffffu);
      Blds[buf][bn + 2 * e + 1][bk] = (u16)(vals[e] >> 16);
    }
  };

  auto compute = [&](int buf) {
#pragma unroll
    for (int kc = 0; kc < 2; ++kc) {
      Frag a[4], bfr[2];
#pragma unroll
      for (int i = 0; i < 4; ++i)
        a[i] = load_fragA<64>(Alds[buf], waveM * 64 + i * 16, kc * 32, lane);
#pragma unroll
      for (int j = 0; j < 2; ++j)
        bfr[j] = load_fragB<64>(Blds[buf], waveN * 32 + j * 16, kc * 32, lane);
#pragma unroll
      for (int i = 0; i < 4; ++i)
#pragma unroll
        for (int j = 0; j < 2; ++j)
          acc[i][j] = __builtin_amdgcn_wmma_f32_16x16x32_bf16(
              false, a[i].v, false, bfr[j].v, (short)0, acc[i][j], false, false);
    }
  };

  stage(0, 0);
  int buf = 0;
  for (int k0 = 0; k0 < K; k0 += 64) {
    wait_async0();
    __syncthreads();
    if (k0 + 64 < K) stage(buf ^ 1, k0 + 64);
    compute(buf);
    buf ^= 1;
  }

  const int r0 = (lane >> 4) << 3;
  const int col = lane & 15;
#pragma unroll
  for (int i = 0; i < 4; ++i)
#pragma unroll
    for (int j = 0; j < 2; ++j) {
      const int gn = n0 + waveN * 32 + j * 16 + col;
      const float bv = bias ? bias[gn] : 0.f;
#pragma unroll
      for (int rr = 0; rr < 8; ++rr) {
        const int gm = m0 + waveM * 64 + i * 16 + r0 + rr;
        float v = acc[i][j][rr] + bv;
        if (gelu) v = 0.5f * v * (1.0f + erff(v * 0.70710678118654752f));
        const size_t idx = (size_t)gm * N + gn;
        if (outF) outF[idx] = v;
        if (outB) outB[idx] = f2bf(v);
      }
    }
}

// ---------------------------------------------------------------------------
// Scores: S[bh,l,m] = (Q.K^T)*scale (+ prev in place), local mask applied
// ---------------------------------------------------------------------------
__global__ __launch_bounds__(128) void attn_scores(
    const u16* __restrict__ Q, const u16* __restrict__ Kmat,
    float* __restrict__ S, int add_prev) {
  __shared__ u16 Qs[64][64];
  __shared__ u16 Ks[64][64];  // rows are key index m, cols dk -> B^T layout
  const int t = threadIdx.x, lane = t & 31, w = t >> 5;
  const int wm = w >> 1, wn = w & 1;
  const int bh = blockIdx.z;
  const int l0 = blockIdx.y * 64;
  const int m0 = blockIdx.x * 64;
  {
    const int row = t >> 1, half = t & 1;
    const u16* gq = Q + ((size_t)bh * SEQ + l0 + row) * DKH + half * 32;
    const u16* gk = Kmat + ((size_t)bh * SEQ + m0 + row) * DKH + half * 32;
    u16* lq = &Qs[row][half * 32];
    u16* lk = &Ks[row][half * 32];
#if HAVE_ASYNC_LDS
    cp_g2l_16(lq, gq);      cp_g2l_16(lq + 8, gq + 8);
    cp_g2l_16(lq + 16, gq + 16); cp_g2l_16(lq + 24, gq + 24);
    cp_g2l_16(lk, gk);      cp_g2l_16(lk + 8, gk + 8);
    cp_g2l_16(lk + 16, gk + 16); cp_g2l_16(lk + 24, gk + 24);
#else
    const uint4* gq4 = (const uint4*)gq;
    const uint4* gk4 = (const uint4*)gk;
    uint4 a0 = gq4[0], a1 = gq4[1], a2 = gq4[2], a3 = gq4[3];
    uint4 b0 = gk4[0], b1 = gk4[1], b2 = gk4[2], b3 = gk4[3];
    uint4* lq4 = (uint4*)lq;
    uint4* lk4 = (uint4*)lk;
    lq4[0] = a0; lq4[1] = a1; lq4[2] = a2; lq4[3] = a3;
    lk4[0] = b0; lk4[1] = b1; lk4[2] = b2; lk4[3] = b3;
#endif
  }
  wait_async0();
  __syncthreads();
  const v8f vzero = {0.f, 0.f, 0.f, 0.f, 0.f, 0.f, 0.f, 0.f};
  v8f acc[2][2] = {{vzero, vzero}, {vzero, vzero}};
#pragma unroll
  for (int kc = 0; kc < 2; ++kc) {
    Frag a0 = load_fragA<64>(Qs, wm * 32, kc * 32, lane);
    Frag a1 = load_fragA<64>(Qs, wm * 32 + 16, kc * 32, lane);
    Frag b0 = load_fragB<64>(Ks, wn * 32, kc * 32, lane);
    Frag b1 = load_fragB<64>(Ks, wn * 32 + 16, kc * 32, lane);
    acc[0][0] = __builtin_amdgcn_wmma_f32_16x16x32_bf16(false, a0.v, false, b0.v,
                                                        (short)0, acc[0][0], false, false);
    acc[0][1] = __builtin_amdgcn_wmma_f32_16x16x32_bf16(false, a0.v, false, b1.v,
                                                        (short)0, acc[0][1], false, false);
    acc[1][0] = __builtin_amdgcn_wmma_f32_16x16x32_bf16(false, a1.v, false, b0.v,
                                                        (short)0, acc[1][0], false, false);
    acc[1][1] = __builtin_amdgcn_wmma_f32_16x16x32_bf16(false, a1.v, false, b1.v,
                                                        (short)0, acc[1][1], false, false);
  }
  const int r0 = (lane >> 4) << 3;
  const int col = lane & 15;
#pragma unroll
  for (int i = 0; i < 2; ++i)
#pragma unroll
    for (int j = 0; j < 2; ++j)
#pragma unroll
      for (int rr = 0; rr < 8; ++rr) {
        const int gl = l0 + wm * 32 + i * 16 + r0 + rr;
        const int gm = m0 + wn * 32 + j * 16 + col;
        const size_t idx = ((size_t)bh * SEQ + gl) * SEQ + gm;
        float s = acc[i][j][rr] * SCL;
        if (add_prev) s += S[idx];
        int dd = gl - gm;
        if (dd < 0) dd = -dd;
        if (dd < HWIN) s = NEGV;
        S[idx] = s;
      }
}

// ---------------------------------------------------------------------------
// Softmax over rows of S (length 1024), one wave per row, bf16 output P
// ---------------------------------------------------------------------------
__global__ __launch_bounds__(128) void softmax_rows(const float* __restrict__ S,
                                                    u16* __restrict__ P) {
  const int row = blockIdx.x * 4 + (threadIdx.x >> 5);
  const int lane = threadIdx.x & 31;
  const float* sp = S + (size_t)row * SEQ;
  float sv[32];
  float m = -3.4e38f;
#pragma unroll
  for (int c = 0; c < 32; ++c) {
    sv[c] = sp[lane + 32 * c];
    m = fmaxf(m, sv[c]);
  }
#pragma unroll
  for (int off = 16; off > 0; off >>= 1) m = fmaxf(m, __shfl_xor(m, off, 32));
  float sum = 0.f;
#pragma unroll
  for (int c = 0; c < 32; ++c) {
    sv[c] = __expf(sv[c] - m);
    sum += sv[c];
  }
#pragma unroll
  for (int off = 16; off > 0; off >>= 1) sum += __shfl_xor(sum, off, 32);
  const float inv = 1.0f / sum;
  u16* pp = P + (size_t)row * SEQ;
#pragma unroll
  for (int c = 0; c < 32; ++c) pp[lane + 32 * c] = f2bf(sv[c] * inv);
}

// ---------------------------------------------------------------------------
// O[bh,l,dk] = P[bh,l,:] @ V[bh,:,dk]
// ---------------------------------------------------------------------------
__global__ __launch_bounds__(128) void attn_pv(const u16* __restrict__ P,
                                               const u16* __restrict__ V,
                                               float* __restrict__ O) {
  __shared__ u16 Ps[64][32];
  __shared__ u16 Vt[64][32];  // [n=dk][k=m]
  const int t = threadIdx.x, lane = t & 31, w = t >> 5;
  const int wm = w >> 1, wn = w & 1;
  const int bh = blockIdx.y;
  const int l0 = blockIdx.x * 64;
  const v8f vzero = {0.f, 0.f, 0.f, 0.f, 0.f, 0.f, 0.f, 0.f};
  v8f acc[2][2] = {{vzero, vzero}, {vzero, vzero}};
  const int prow = t >> 1, phalf = t & 1;
  const int vk = t >> 2, vn = (t & 3) * 16;

  for (int m0 = 0; m0 < SEQ; m0 += 32) {
    __syncthreads();
    const u16* gp = P + ((size_t)bh * SEQ + l0 + prow) * SEQ + m0 + phalf * 16;
    u16* lp = &Ps[prow][phalf * 16];
#if HAVE_ASYNC_LDS
    cp_g2l_16(lp, gp);
    cp_g2l_16(lp + 8, gp + 8);
#else
    {
      const uint4* gp4 = (const uint4*)gp;
      uint4 p0 = gp4[0], p1 = gp4[1];
      uint4* lp4 = (uint4*)lp;
      lp4[0] = p0; lp4[1] = p1;
    }
#endif
    const uint4* gv = (const uint4*)(V + ((size_t)bh * SEQ + m0 + vk) * DKH + vn);
    uint4 q0 = gv[0], q1 = gv[1];
    u32 vals[8] = {q0.x, q0.y, q0.z, q0.w, q1.x, q1.y, q1.z, q1.w};
#pragma unroll
    for (int e = 0; e < 8; ++e) {
      Vt[vn + 2 * e][vk]     = (u16)(vals[e] & 0xffffu);
      Vt[vn + 2 * e + 1][vk] = (u16)(vals[e] >> 16);
    }
    wait_async0();
    __syncthreads();
    Frag a0 = load_fragA<32>(Ps, wm * 32, 0, lane);
    Frag a1 = load_fragA<32>(Ps, wm * 32 + 16, 0, lane);
    Frag b0 = load_fragB<32>(Vt, wn * 32, 0, lane);
    Frag b1 = load_fragB<32>(Vt, wn * 32 + 16, 0, lane);
    acc[0][0] = __builtin_amdgcn_wmma_f32_16x16x32_bf16(false, a0.v, false, b0.v,
                                                        (short)0, acc[0][0], false, false);
    acc[0][1] = __builtin_amdgcn_wmma_f32_16x16x32_bf16(false, a0.v, false, b1.v,
                                                        (short)0, acc[0][1], false, false);
    acc[1][0] = __builtin_amdgcn_wmma_f32_16x16x32_bf16(false, a1.v, false, b0.v,
                                                        (short)0, acc[1][0], false, false);
    acc[1][1] = __builtin_amdgcn_wmma_f32_16x16x32_bf16(false, a1.v, false, b1.v,
                                                        (short)0, acc[1][1], false, false);
  }
  const int r0 = (lane >> 4) << 3;
  const int col = lane & 15;
#pragma unroll
  for (int i = 0; i < 2; ++i)
#pragma unroll
    for (int j = 0; j < 2; ++j)
#pragma unroll
      for (int rr = 0; rr < 8; ++rr) {
        const int gl = l0 + wm * 32 + i * 16 + r0 + rr;
        const int gd = wn * 32 + j * 16 + col;
        O[((size_t)bh * SEQ + gl) * DKH + gd] = acc[i][j][rr];
      }
}

// ---------------------------------------------------------------------------
// Elementwise helpers
// ---------------------------------------------------------------------------
__global__ void conv_f32_bf16(const float* __restrict__ in, u16* __restrict__ out,
                              int n) {
  int i = blockIdx.x * blockDim.x + threadIdx.x;
  if (i < n) out[i] = f2bf(in[i]);
}

__global__ void perm_blhd_to_bhld(const float* __restrict__ in,
                                  u16* __restrict__ out, int n) {
  int i = blockIdx.x * blockDim.x + threadIdx.x;
  if (i >= n) return;
  int d = i % DKH;
  int h = (i / DKH) % NH;
  int l = (i / (DKH * NH)) % SEQ;
  int b = i / (DKH * NH * SEQ);
  out[(((size_t)(b * NH + h) * SEQ) + l) * DKH + d] = f2bf(in[i]);
}

__global__ void perm_bhld_to_blhd(const float* __restrict__ in,
                                  u16* __restrict__ out, int n) {
  int i = blockIdx.x * blockDim.x + threadIdx.x;
  if (i >= n) return;
  int d = i % DKH;
  int l = (i / DKH) % SEQ;
  int h = (i / (DKH * SEQ)) % NH;
  int b = i / (DKH * SEQ * NH);
  out[(((size_t)(b * SEQ + l) * NH) + h) * DKH + d] = f2bf(in[i]);
}

// x = LN(x + y) ; writes fp32 state in place + bf16 copy
__global__ __launch_bounds__(256) void residual_ln(
    float* __restrict__ X, const float* __restrict__ Y,
    const float* __restrict__ g, const float* __restrict__ be,
    u16* __restrict__ Xb) {
  const int row = blockIdx.x * 8 + (threadIdx.x >> 5);
  const int lane = threadIdx.x & 31;
  float* xp = X + (size_t)row * DIM;
  const float* yp = Y + (size_t)row * DIM;
  float vals[16];
  float s = 0.f;
#pragma unroll
  for (int c = 0; c < 16; ++c) {
    vals[c] = xp[lane + 32 * c] + yp[lane + 32 * c];
    s += vals[c];
  }
#pragma unroll
  for (int off = 16; off > 0; off >>= 1) s += __shfl_xor(s, off, 32);
  const float mean = s * (1.0f / DIM);
  float vs = 0.f;
#pragma unroll
  for (int c = 0; c < 16; ++c) {
    float d = vals[c] - mean;
    vs += d * d;
  }
#pragma unroll
  for (int off = 16; off > 0; off >>= 1) vs += __shfl_xor(vs, off, 32);
  const float rstd = rsqrtf(vs * (1.0f / DIM) + 1e-5f);
#pragma unroll
  for (int c = 0; c < 16; ++c) {
    const int col = lane + 32 * c;
    const float o = (vals[c] - mean) * rstd * g[col] + be[col];
    xp[col] = o;
    Xb[(size_t)row * DIM + col] = f2bf(o);
  }
}

// ---------------------------------------------------------------------------
// Host orchestration
// ---------------------------------------------------------------------------
extern "C" void kernel_launch(void* const* d_in, const int* in_sizes, int n_in,
                              void* d_out, int out_size, void* d_ws,
                              size_t ws_size, hipStream_t stream) {
  (void)in_sizes; (void)n_in; (void)out_size; (void)ws_size;
  const float* src  = (const float*)d_in[0];
  const float* Wq   = (const float*)d_in[1];
  const float* bq   = (const float*)d_in[2];
  const float* Wk   = (const float*)d_in[3];
  const float* bk   = (const float*)d_in[4];
  const float* Wv   = (const float*)d_in[5];
  const float* bv   = (const float*)d_in[6];
  const float* Wo   = (const float*)d_in[7];
  const float* bo   = (const float*)d_in[8];
  const float* W1   = (const float*)d_in[9];
  const float* b1   = (const float*)d_in[10];
  const float* W2   = (const float*)d_in[11];
  const float* b2   = (const float*)d_in[12];
  const float* ln1g = (const float*)d_in[13];
  const float* ln1b = (const float*)d_in[14];
  const float* ln2g = (const float*)d_in[15];
  const float* ln2b = (const float*)d_in[16];

  char* ws = (char*)d_ws;
  size_t off = 0;
  auto a256 = [](size_t x) { return (x + 255) & ~(size_t)255; };
  float* S   = (float*)(ws + off); off = a256(off + (size_t)BH * SEQ * SEQ * 4);
  u16*   P   = (u16*)(ws + off);   off = a256(off + (size_t)BH * SEQ * SEQ * 2);
  u16*   qbf = (u16*)(ws + off);   off = a256(off + (size_t)BH * SEQ * DKH * 2);
  u16*   kbf = (u16*)(ws + off);   off = a256(off + (size_t)BH * SEQ * DKH * 2);
  u16*   vbf = (u16*)(ws + off);   off = a256(off + (size_t)BH * SEQ * DKH * 2);
  float* ao  = (float*)(ws + off); off = a256(off + (size_t)BH * SEQ * DKH * 4);
  u16*   aob = (u16*)(ws + off);   off = a256(off + (size_t)BLM * DIM * 2);
  float* x   = (float*)(ws + off); off = a256(off + (size_t)BLM * DIM * 4);
  u16*   xb  = (u16*)(ws + off);   off = a256(off + (size_t)BLM * DIM * 2);
  float* tmp = (float*)(ws + off); off = a256(off + (size_t)BLM * DIM * 4);
  u16*   hb  = (u16*)(ws + off);   off = a256(off + (size_t)BLM * DFF * 2);
  u16*   wqb = (u16*)(ws + off);   off = a256(off + (size_t)NLAY * DIM * DIM * 2);
  u16*   wkb = (u16*)(ws + off);   off = a256(off + (size_t)NLAY * DIM * DIM * 2);
  u16*   wvb = (u16*)(ws + off);   off = a256(off + (size_t)NLAY * DIM * DIM * 2);
  u16*   wob = (u16*)(ws + off);   off = a256(off + (size_t)NLAY * DIM * DIM * 2);
  u16*   w1b = (u16*)(ws + off);   off = a256(off + (size_t)NLAY * DIM * DFF * 2);
  u16*   w2b = (u16*)(ws + off);   off = a256(off + (size_t)NLAY * DFF * DIM * 2);

  const int CT = 256;
  const int nww = NLAY * DIM * DIM;
  const int nwf = NLAY * DIM * DFF;
  conv_f32_bf16<<<(nww + CT - 1) / CT, CT, 0, stream>>>(Wq, wqb, nww);
  conv_f32_bf16<<<(nww + CT - 1) / CT, CT, 0, stream>>>(Wk, wkb, nww);
  conv_f32_bf16<<<(nww + CT - 1) / CT, CT, 0, stream>>>(Wv, wvb, nww);
  conv_f32_bf16<<<(nww + CT - 1) / CT, CT, 0, stream>>>(Wo, wob, nww);
  conv_f32_bf16<<<(nwf + CT - 1) / CT, CT, 0, stream>>>(W1, w1b, nwf);
  conv_f32_bf16<<<(nwf + CT - 1) / CT, CT, 0, stream>>>(W2, w2b, nwf);

  (void)hipMemcpyAsync(x, src, (size_t)BLM * DIM * 4, hipMemcpyDeviceToDevice,
                       stream);
  conv_f32_bf16<<<(BLM * DIM + CT - 1) / CT, CT, 0, stream>>>(src, xb, BLM * DIM);

  const dim3 gD(DIM / 128, BLM / 128);
  const dim3 gF(DFF / 128, BLM / 128);
  const int nact = BLM * DIM;

  for (int l = 0; l < NLAY; ++l) {
    // Q, K, V projections + head permute to [B,H,L,DK] bf16
    gemm_bf16<<<gD, 256, 0, stream>>>(xb, wqb + (size_t)l * DIM * DIM,
                                      bq + l * DIM, tmp, nullptr, BLM, DIM, DIM, 0);
    perm_blhd_to_bhld<<<(nact + CT - 1) / CT, CT, 0, stream>>>(tmp, qbf, nact);
    gemm_bf16<<<gD, 256, 0, stream>>>(xb, wkb + (size_t)l * DIM * DIM,
                                      bk + l * DIM, tmp, nullptr, BLM, DIM, DIM, 0);
    perm_blhd_to_bhld<<<(nact + CT - 1) / CT, CT, 0, stream>>>(tmp, kbf, nact);
    gemm_bf16<<<gD, 256, 0, stream>>>(xb, wvb + (size_t)l * DIM * DIM,
                                      bv + l * DIM, tmp, nullptr, BLM, DIM, DIM, 0);
    perm_blhd_to_bhld<<<(nact + CT - 1) / CT, CT, 0, stream>>>(tmp, vbf, nact);

    // scores (+prev in place), mask; softmax; P@V
    attn_scores<<<dim3(SEQ / 64, SEQ / 64, BH), 128, 0, stream>>>(qbf, kbf, S,
                                                                  l > 0 ? 1 : 0);
    softmax_rows<<<(BH * SEQ) / 4, 128, 0, stream>>>(S, P);
    attn_pv<<<dim3(SEQ / 64, BH), 128, 0, stream>>>(P, vbf, ao);
    perm_bhld_to_blhd<<<(nact + CT - 1) / CT, CT, 0, stream>>>(ao, aob, nact);

    // output projection, residual + LN1
    gemm_bf16<<<gD, 256, 0, stream>>>(aob, wob + (size_t)l * DIM * DIM,
                                      bo + l * DIM, tmp, nullptr, BLM, DIM, DIM, 0);
    residual_ln<<<BLM / 8, 256, 0, stream>>>(x, tmp, ln1g + l * DIM,
                                             ln1b + l * DIM, xb);

    // FFN: GELU(x@W1+b1)@W2+b2, residual + LN2
    gemm_bf16<<<gF, 256, 0, stream>>>(xb, w1b + (size_t)l * DIM * DFF,
                                      b1 + l * DFF, nullptr, hb, BLM, DFF, DIM, 1);
    gemm_bf16<<<gD, 256, 0, stream>>>(hb, w2b + (size_t)l * DFF * DIM,
                                      b2 + l * DIM, tmp, nullptr, BLM, DIM, DFF, 0);
    residual_ln<<<BLM / 8, 256, 0, stream>>>(x, tmp, ln2g + l * DIM,
                                             ln2b + l * DIM, xb);
  }

  (void)hipMemcpyAsync(d_out, x, (size_t)BLM * DIM * 4, hipMemcpyDeviceToDevice,
                       stream);
}
